// TemporalGNN_7421703488006
// MI455X (gfx1250) — matrix operand order repaired and checked by reading, not compiled, not run on hardware
//
#include <hip/hip_runtime.h>

typedef __attribute__((ext_vector_type(16))) __bf16          v16bf;
typedef __attribute__((ext_vector_type(8)))  float           v8f;

#define DEV __device__ __forceinline__

DEV unsigned short f2bf(float f) {          // round-to-nearest-even f32 -> bf16
  unsigned int u = __float_as_uint(f);
  u += 0x7FFFu + ((u >> 16) & 1u);
  return (unsigned short)(u >> 16);
}

// 4 consecutive fp32 RMWs at device scope, issued as one hardware clause.
DEV void atom_add_f32x4(float* p, float a, float b, float c, float d) {
  asm volatile(
      "s_clause 0x3\n\t"
      "global_atomic_add_f32 %0, %1, off scope:SCOPE_DEV\n\t"
      "global_atomic_add_f32 %0, %2, off offset:4 scope:SCOPE_DEV\n\t"
      "global_atomic_add_f32 %0, %3, off offset:8 scope:SCOPE_DEV\n\t"
      "global_atomic_add_f32 %0, %4, off offset:12 scope:SCOPE_DEV"
      :: "v"(p), "v"(a), "v"(b), "v"(c), "v"(d) : "memory");
}

DEV void atom_add_f32(float* p, float v) {
  asm volatile("global_atomic_add_f32 %0, %1, off scope:SCOPE_DEV"
               :: "v"(p), "v"(v) : "memory");
}

// ---------------------------------------------------------------------------
// Edge-parallel scatter-add: out[dst] += norm * h[src].
// One thread = one (edge, 4-feature chunk); 32-bit index math throughout.
// ---------------------------------------------------------------------------
extern "C" __global__ void __launch_bounds__(256)
k_scatter(const float* __restrict__ h, const long long* __restrict__ src,
          const long long* __restrict__ dst, const float* __restrict__ nrm,
          float* __restrict__ out, int fsh /* F/4 == 1<<fsh */, unsigned total) {
  unsigned tid = blockIdx.x * 256u + threadIdx.x;
  if (tid >= total) return;
  unsigned e = tid >> fsh;
  unsigned c = tid & ((1u << fsh) - 1u);
  int fsh2 = fsh + 2;                       // F == 4 << fsh
  unsigned s = (unsigned)src[e];
  unsigned d = (unsigned)dst[e];
  float w = nrm[e];
  const float4 v = *(const float4*)(h + ((size_t)s << fsh2) + c * 4u);
  float* o = out + ((size_t)d << fsh2) + c * 4u;
  atom_add_f32x4(o, v.x * w, v.y * w, v.z * w, v.w * w);
}

// ---------------------------------------------------------------------------
// WMMA GEMM:  C[M,N] = A[M,K] (f32 -> bf16 in regs) * Wpk (pre-packed bf16).
// 64-thread blocks (2 waves); each wave owns 4 consecutive 16x16 N-tiles of
// one M-tile, so one A-fragment feeds 4 v_wmma_f32_16x16x32_bf16 issues.
// Requires M%16==0, N%128==0, K%32==0 (true here: 50000, 256/128, 768/256).
// ---------------------------------------------------------------------------
extern "C" __global__ void __launch_bounds__(64)
k_gemm_bf16(const float* __restrict__ A, const unsigned short* __restrict__ Wpk,
            float* __restrict__ C, int N, int K) {
  const int lane = threadIdx.x & 31;
  const int wave = threadIdx.x >> 5;
  const int nt0  = (blockIdx.x * 2 + wave) * 4;   // first of 4 N-tiles
  const int mt   = blockIdx.y;
  const int half = lane >> 4;
  const int mrow = mt * 16 + (lane & 15);
  const float* arow = A + (size_t)mrow * K + half * 8;
  const int ntiles = N >> 4;
  v8f acc0 = {}, acc1 = {}, acc2 = {}, acc3 = {};
  for (int kt = 0; kt < (K >> 5); ++kt) {
    const float* ap = arow + (kt << 5);
    float4 a0 = *(const float4*)(ap + 0);
    float4 a1 = *(const float4*)(ap + 4);
    float4 a2 = *(const float4*)(ap + 16);
    float4 a3 = *(const float4*)(ap + 20);
    // native casts: paired fptrunc should select the hardware pack-convert
    v16bf af;
    af[0]  = (__bf16)a0.x; af[1]  = (__bf16)a0.y;
    af[2]  = (__bf16)a0.z; af[3]  = (__bf16)a0.w;
    af[4]  = (__bf16)a1.x; af[5]  = (__bf16)a1.y;
    af[6]  = (__bf16)a1.z; af[7]  = (__bf16)a1.w;
    af[8]  = (__bf16)a2.x; af[9]  = (__bf16)a2.y;
    af[10] = (__bf16)a2.z; af[11] = (__bf16)a2.w;
    af[12] = (__bf16)a3.x; af[13] = (__bf16)a3.y;
    af[14] = (__bf16)a3.z; af[15] = (__bf16)a3.w;
    const v16bf* b = (const v16bf*)Wpk + ((size_t)kt * ntiles + nt0) * 32 + lane;
    v16bf b0 = b[0];
    v16bf b1 = b[32];
    v16bf b2 = b[64];
    v16bf b3 = b[96];
    acc0 = __builtin_amdgcn_wmma_f32_16x16x32_bf16(false, af, false, b0, (short)0, acc0, false, false);
    acc1 = __builtin_amdgcn_wmma_f32_16x16x32_bf16(false, af, false, b1, (short)0, acc1, false, false);
    acc2 = __builtin_amdgcn_wmma_f32_16x16x32_bf16(false, af, false, b2, (short)0, acc2, false, false);
    acc3 = __builtin_amdgcn_wmma_f32_16x16x32_bf16(false, af, false, b3, (short)0, acc3, false, false);
  }
  // store 4 tiles; rows rbase..rbase+7, no guards (M%16==0)
  const int rbase = mt * 16 + half * 8;
  float* cp = C + (size_t)rbase * N + (nt0 * 16 + (lane & 15));
#pragma unroll
  for (int v = 0; v < 8; ++v) {
    float* r = cp + (size_t)v * N;
    r[0]  = acc0[v];
    r[16] = acc1[v];
    r[32] = acc2[v];
    r[48] = acc3[v];
  }
}

// Pack f32 weight [K,N] into per-lane-contiguous bf16 WMMA B fragments.
// Tile t = kt*(N/16)+nt; element (t, lane, i) = W[kt*32+(lane<16?0:8)+(i&7)+(i>>3)*16][nt*16+(lane&15)]
extern "C" __global__ void __launch_bounds__(256)
k_pack_w(const float* __restrict__ W, unsigned short* __restrict__ Wpk, int K, int N) {
  unsigned tid = blockIdx.x * 256u + threadIdx.x;
  if (tid >= (unsigned)(K * N)) return;
  int i    = (int)(tid & 15);
  int lane = (int)((tid >> 4) & 31);
  int t    = (int)(tid >> 9);
  int ntiles = N >> 4;
  int kt = t / ntiles, nt = t - kt * ntiles;
  int k = kt * 32 + ((lane < 16) ? 0 : 8) + (i & 7) + ((i >> 3) << 4);
  int n = nt * 16 + (lane & 15);
  Wpk[tid] = f2bf(W[(size_t)k * N + n]);
}

extern "C" __global__ void __launch_bounds__(256)
k_fill(float* __restrict__ p, float val, unsigned n) {
  unsigned i = blockIdx.x * 256u + threadIdx.x;
  if (i < n) p[i] = val;
}

extern "C" __global__ void __launch_bounds__(256)
k_fill4(float4* __restrict__ p, float val, unsigned n4) {
  unsigned i = blockIdx.x * 256u + threadIdx.x;
  if (i < n4) p[i] = make_float4(val, val, val, val);
}

extern "C" __global__ void __launch_bounds__(256)
k_deg(const long long* __restrict__ dst, const float* __restrict__ ew,
      float* __restrict__ deg, unsigned E) {
  unsigned e = blockIdx.x * 256u + threadIdx.x;
  if (e < E) atom_add_f32(&deg[(unsigned)dst[e]], ew[e]);
}

extern "C" __global__ void __launch_bounds__(256)
k_dinv(float* __restrict__ p, unsigned n) {   // in-place deg -> deg^{-1/2}
  unsigned i = blockIdx.x * 256u + threadIdx.x;
  if (i < n) { float d = p[i]; p[i] = (d > 0.f) ? rsqrtf(d) : 0.f; }
}

extern "C" __global__ void __launch_bounds__(256)
k_norm(const long long* __restrict__ src, const long long* __restrict__ dst,
       const float* __restrict__ ew, const float* __restrict__ dinv,
       float* __restrict__ nrm, unsigned E) {
  unsigned e = blockIdx.x * 256u + threadIdx.x;
  if (e < E) nrm[e] = dinv[(unsigned)src[e]] * ew[e] * dinv[(unsigned)dst[e]];
}

// epilogue: agg + self-loop + bias, then eval-mode BN, then ReLU (float4)
extern "C" __global__ void __launch_bounds__(256)
k_fuse_bn_relu(const float* __restrict__ agg, const float* __restrict__ h,
               const float* __restrict__ dinv, const float* __restrict__ bias,
               const float* __restrict__ gamma, const float* __restrict__ beta,
               const float* __restrict__ mean, const float* __restrict__ var,
               float* __restrict__ out, int fsh /* F == 1<<fsh */, unsigned total4) {
  unsigned tid = blockIdx.x * 256u + threadIdx.x;
  if (tid >= total4) return;
  unsigned base = tid * 4u;
  unsigned i = base >> fsh;
  unsigned f = base & ((1u << fsh) - 1u);
  float di = dinv[i];
  float d2 = di * di;
  float4 ag = *(const float4*)(agg + base);
  float4 hh = *(const float4*)(h + base);
  float4 bs = *(const float4*)(bias + f);
  float4 mn = *(const float4*)(mean + f);
  float4 vr = *(const float4*)(var + f);
  float4 gm = *(const float4*)(gamma + f);
  float4 bt = *(const float4*)(beta + f);
  float4 o;
  o.x = fmaxf((ag.x + d2 * hh.x + bs.x - mn.x) * rsqrtf(vr.x + 1e-5f) * gm.x + bt.x, 0.f);
  o.y = fmaxf((ag.y + d2 * hh.y + bs.y - mn.y) * rsqrtf(vr.y + 1e-5f) * gm.y + bt.y, 0.f);
  o.z = fmaxf((ag.z + d2 * hh.z + bs.z - mn.z) * rsqrtf(vr.z + 1e-5f) * gm.z + bt.z, 0.f);
  o.w = fmaxf((ag.w + d2 * hh.w + bs.w - mn.w) * rsqrtf(vr.w + 1e-5f) * gm.w + bt.w, 0.f);
  *(float4*)(out + base) = o;
}

// final layer epilogue: agg + self-loop + bias, then L2 row-normalize
extern "C" __global__ void
k_final_norm(const float* __restrict__ agg, const float* __restrict__ h,
             const float* __restrict__ dinv, const float* __restrict__ bias,
             float* __restrict__ out) {
  __shared__ float red[128];
  const int i = blockIdx.x, f = threadIdx.x, F = blockDim.x;
  unsigned idx = (unsigned)i * F + f;
  float di = dinv[i];
  float v = agg[idx] + di * di * h[idx] + bias[f];
  red[f] = v * v;
  __syncthreads();
  for (int s = F >> 1; s > 0; s >>= 1) {
    if (f < s) red[f] += red[f + s];
    __syncthreads();
  }
  float nr = fmaxf(sqrtf(red[0]), 1e-12f);
  out[idx] = v / nr;
}

// ---------------------------------------------------------------------------
static inline unsigned nblk(long long total, int bs) {
  return (unsigned)((total + bs - 1) / bs);
}

extern "C" void kernel_launch(void* const* d_in, const int* in_sizes, int n_in,
                              void* d_out, int out_size, void* d_ws, size_t ws_size,
                              hipStream_t stream) {
  const float*     x   = (const float*)d_in[0];
  const long long* ei  = (const long long*)d_in[1];   // int64 [2,E]
  const float*     ew  = (const float*)d_in[2];
  const float*     W1  = (const float*)d_in[3];
  const float*     b1  = (const float*)d_in[4];
  const float*     W2  = (const float*)d_in[5];
  const float*     b2  = (const float*)d_in[6];
  const float*     W3  = (const float*)d_in[7];
  const float*     b3  = (const float*)d_in[8];
  const float*     g1  = (const float*)d_in[9];
  const float*     be1 = (const float*)d_in[10];
  const float*     m1  = (const float*)d_in[11];
  const float*     v1  = (const float*)d_in[12];
  const float*     g2  = (const float*)d_in[13];
  const float*     be2 = (const float*)d_in[14];
  const float*     m2  = (const float*)d_in[15];
  const float*     v2  = (const float*)d_in[16];

  const long long E   = in_sizes[2];
  const int       HID = in_sizes[4];
  const int       IN  = in_sizes[3] / HID;
  const int       OUT = in_sizes[8];
  const long long Nn  = (long long)in_sizes[0] / IN;

  const long long* srcI = ei;
  const long long* dstI = ei + E;

  // workspace carve (256B aligned)
  char* ws = (char*)d_ws;
  size_t off = 0;
  auto carve = [&](size_t bytes) -> char* {
    char* p = ws + off;
    off += (bytes + 255) & ~(size_t)255;
    return p;
  };
  float* actA = (float*)carve((size_t)Nn * HID * 4);   // activations (GEMM input)
  float* hB   = (float*)carve((size_t)Nn * HID * 4);   // GEMM output
  float* aggC = (float*)carve((size_t)Nn * HID * 4);   // scatter accumulator
  float* dinv = (float*)carve((size_t)Nn * 4);         // deg -> deg^-1/2
  float* nrmE = (float*)carve((size_t)E * 4);
  unsigned short* w1pk = (unsigned short*)carve((size_t)IN * HID * 2);
  unsigned short* w2pk = (unsigned short*)carve((size_t)HID * HID * 2);
  unsigned short* w3pk = (unsigned short*)carve((size_t)HID * OUT * 2);

  const int fshH = __builtin_ctz((unsigned)(HID / 4));  // 256 -> 6
  const int fshO = __builtin_ctz((unsigned)(OUT / 4));  // 128 -> 5
  const int fshF = __builtin_ctz((unsigned)HID);        // 256 -> 8

  // weight packing (tiny; deterministic each call)
  k_pack_w<<<nblk((long long)IN  * HID, 256), 256, 0, stream>>>(W1, w1pk, IN,  HID);
  k_pack_w<<<nblk((long long)HID * HID, 256), 256, 0, stream>>>(W2, w2pk, HID, HID);
  k_pack_w<<<nblk((long long)HID * OUT, 256), 256, 0, stream>>>(W3, w3pk, HID, OUT);

  // GCN normalization: deg = 1 (self loop) + scatter(ew); dinv = rsqrt; edge norm
  k_fill<<<nblk(Nn, 256), 256, 0, stream>>>(dinv, 1.0f, (unsigned)Nn);
  k_deg <<<nblk(E, 256), 256, 0, stream>>>(dstI, ew, dinv, (unsigned)E);
  k_dinv<<<nblk(Nn, 256), 256, 0, stream>>>(dinv, (unsigned)Nn);
  k_norm<<<nblk(E, 256), 256, 0, stream>>>(srcI, dstI, ew, dinv, nrmE, (unsigned)E);

  const unsigned nh4 = (unsigned)(Nn * HID / 4);
  const unsigned no4 = (unsigned)(Nn * OUT / 4);

  // ---- layer 1 ----
  k_gemm_bf16<<<dim3(HID / 128, (unsigned)(Nn / 16)), 64, 0, stream>>>(x, w1pk, hB, HID, IN);
  k_fill4<<<nblk(nh4, 256), 256, 0, stream>>>((float4*)aggC, 0.0f, nh4);
  k_scatter<<<nblk(E << fshH, 256), 256, 0, stream>>>(hB, srcI, dstI, nrmE, aggC,
                                                      fshH, (unsigned)(E << fshH));
  k_fuse_bn_relu<<<nblk(nh4, 256), 256, 0, stream>>>(
      aggC, hB, dinv, b1, g1, be1, m1, v1, actA, fshF, nh4);

  // ---- layer 2 ----
  k_gemm_bf16<<<dim3(HID / 128, (unsigned)(Nn / 16)), 64, 0, stream>>>(actA, w2pk, hB, HID, HID);
  k_fill4<<<nblk(nh4, 256), 256, 0, stream>>>((float4*)aggC, 0.0f, nh4);
  k_scatter<<<nblk(E << fshH, 256), 256, 0, stream>>>(hB, srcI, dstI, nrmE, aggC,
                                                      fshH, (unsigned)(E << fshH));
  k_fuse_bn_relu<<<nblk(nh4, 256), 256, 0, stream>>>(
      aggC, hB, dinv, b2, g2, be2, m2, v2, actA, fshF, nh4);

  // ---- layer 3 + L2 normalize ----
  k_gemm_bf16<<<dim3(OUT / 128, (unsigned)(Nn / 16)), 64, 0, stream>>>(actA, w3pk, hB, OUT, HID);
  k_fill4<<<nblk(no4, 256), 256, 0, stream>>>((float4*)aggC, 0.0f, no4);
  k_scatter<<<nblk(E << fshO, 256), 256, 0, stream>>>(hB, srcI, dstI, nrmE, aggC,
                                                      fshO, (unsigned)(E << fshO));
  k_final_norm<<<(unsigned)Nn, OUT, 0, stream>>>(aggC, hB, dinv, b3, (float*)d_out);
}
